// InVAE_25486335935240
// MI455X (gfx1250) — matrix operand order repaired and checked by reading, not compiled, not run on hardware
//
#include <hip/hip_runtime.h>

#define NN 100       // nodes
#define NB 128       // batch

typedef __attribute__((ext_vector_type(16))) _Float16 v16h;
typedef __attribute__((ext_vector_type(8)))  float    v8f;

// ---------------------------------------------------------------- WMMA helpers

static __device__ inline v8f wmma32(v16h a, v16h b, v8f c) {
  // D(16x16,f32) = A(16x32,f16) * B(32x16,f16) + C
  return __builtin_amdgcn_wmma_f32_16x16x32_f16(false, a, false, b, (short)0, c,
                                                false, false);
}

// A fragment (16x32 f16, M=lane&15). rowp = per-lane pointer to row M, stride-1 K.
// Lanes 0-15 hold K {kb+0..15}; lanes 16-31 hold K {kb+16..31}.
static __device__ inline v16h load_a_row(const _Float16* __restrict__ rowp, int kb) {
  const int koff = ((threadIdx.x & 31) >> 4) * 16;
  v16h a;
#pragma unroll
  for (int i = 0; i < 16; ++i) a[i] = rowp[kb + koff + i];
  return a;
}

// B fragment (32x16 f16) from f16 weights stored row-major [Kpad][Nld] (B = W^T).
// Lanes 0-15: N=lane, K = k0+0..15 ; lanes 16-31: N=lane-16, K = k0+16..31.
static __device__ inline v16h load_b_frag(const _Float16* __restrict__ W, int Nld,
                                          int k0, int n0) {
  const int lane = threadIdx.x & 31;
  const int n  = n0 + (lane & 15);
  const int kk = k0 + ((lane >> 4) ? 16 : 0);
  v16h b;
#pragma unroll
  for (int i = 0; i < 16; ++i) b[i] = W[(size_t)(kk + i) * Nld + n];
  return b;
}

// C fragment broadcast of bias[n] (guarded vs padded columns).
static __device__ inline v8f bias_frag(const float* __restrict__ bias, int n0, int OUT) {
  const int n = n0 + (threadIdx.x & 15);
  const float bv = (n < OUT) ? bias[n] : 0.f;
  v8f c;
#pragma unroll
  for (int v = 0; v < 8; ++v) c[v] = bv;
  return c;
}

// Store D fragment (relu) into LDS tile [16][64] row-major f16 (next layer's A).
// D layout: lane<16 -> col=lane, M=v ; lane>=16 -> col=lane-16, M=8+v.
static __device__ inline void store_d_relu(_Float16* __restrict__ t, int n0, v8f d) {
  const int lane = threadIdx.x & 31;
  const int n  = n0 + (lane & 15);
  const int mb = (lane >> 4) * 8;
#pragma unroll
  for (int v = 0; v < 8; ++v) {
    float x = fmaxf(d[v], 0.f);
    t[(mb + v) * 64 + n] = (_Float16)x;
  }
}

// ---------------------------------------------------------------- edge kernel
// One block per (receiver r, batch b). 99 edges (senders s != r, contiguous in
// reference RECV order) -> fr1/fr2/fr3 MLP via WMMA -> masked column sum = Ebar
// row. Writes fused c[b,r,:] = [ x[b,:,r] | Ebar[b,r,:] | 0-pad ] in f16.
template <int P, int DE, int K1, int CP>
__global__ void __launch_bounds__(128)
edge_kernel(const _Float16* __restrict__ xh,                    // [B,P,NN]
            const _Float16* __restrict__ W1, const float* __restrict__ b1,
            const _Float16* __restrict__ W2, const float* __restrict__ b2,
            const _Float16* __restrict__ W3, const float* __restrict__ b3,
            _Float16* __restrict__ cbuf)                        // [B,NN,CP]
{
  constexpr int KS1 = K1 / 32;
  constexpr int NT3 = (DE + 15) / 16;          // <= 2
  const int b = blockIdx.y, r = blockIdx.x;
  const int w = threadIdx.x >> 5, lane = threadIdx.x & 31;
  const int row = lane & 15, half = lane >> 4;

  __shared__ _Float16 sA[4][16][64];
  __shared__ _Float16 sH[4][16][64];
  __shared__ float    sEbar[4][32];

  float eacc[NT3];
#pragma unroll
  for (int nt = 0; nt < NT3; ++nt) eacc[nt] = 0.f;

  const _Float16* xb = xh + (size_t)b * P * NN;

  for (int it = 0; it < 2; ++it) {
    const int tile  = it * 4 + w;              // 0..7 (tile 7 fully masked)
    const int ebase = tile * 16;
    __syncthreads();                           // sA reuse vs previous iteration

    // ---- build A tile: 16 edges x [recv feats | send feats | 0-pad] ----
    {
      const int e = ebase + row;
      const bool valid = e < NN - 1;
      const int s = (e < r) ? e : e + 1;       // sender skips r
      for (int cc = half * (K1 / 2); cc < (half + 1) * (K1 / 2); ++cc) {
        float v = 0.f;
        if (valid) {
          if (cc < P)            v = (float)xb[cc * NN + r];
          else if (cc < 2 * P)   v = (float)xb[(cc - P) * NN + s];
        }
        sA[w][row][cc] = (_Float16)v;
      }
    }
    __syncthreads();

    // ---- fr1: K1 -> 64 ----
#pragma unroll
    for (int nt = 0; nt < 4; ++nt) {
      v8f acc = bias_frag(b1, nt * 16, 64);
#pragma unroll
      for (int k = 0; k < KS1; ++k)
        acc = wmma32(load_a_row(&sA[w][row][0], k * 32),
                     load_b_frag(W1, 64, k * 32, nt * 16), acc);
      store_d_relu(&sH[w][0][0], nt * 16, acc);
    }
    __syncthreads();

    // ---- fr2: 64 -> 64 ----
#pragma unroll
    for (int nt = 0; nt < 4; ++nt) {
      v8f acc = bias_frag(b2, nt * 16, 64);
#pragma unroll
      for (int k = 0; k < 2; ++k)
        acc = wmma32(load_a_row(&sH[w][row][0], k * 32),
                     load_b_frag(W2, 64, k * 32, nt * 16), acc);
      store_d_relu(&sA[w][0][0], nt * 16, acc);
    }
    __syncthreads();

    // ---- fr3: 64 -> DE, relu, masked column-sum (scatter-add onto receiver) ----
#pragma unroll
    for (int nt = 0; nt < NT3; ++nt) {
      v8f acc = bias_frag(b3, nt * 16, DE);
#pragma unroll
      for (int k = 0; k < 2; ++k)
        acc = wmma32(load_a_row(&sA[w][row][0], k * 32),
                     load_b_frag(W3, NT3 * 16, k * 32, nt * 16), acc);
      float cs = 0.f;
      const int mb = half * 8;
#pragma unroll
      for (int v = 0; v < 8; ++v)
        if (ebase + mb + v < NN - 1) cs += fmaxf(acc[v], 0.f);
      cs += __shfl_xor(cs, 16, 32);            // fold both M-halves of the column
      eacc[nt] += cs;
    }
  }

  if (lane < 16)
#pragma unroll
    for (int nt = 0; nt < NT3; ++nt) sEbar[w][nt * 16 + lane] = eacc[nt];
  __syncthreads();

  // ---- write c[b, r, :]  (x part | Ebar | zero pad) ----
  const int tid = threadIdx.x;
  if (tid < CP) {
    float v;
    if (tid < P) {
      v = (float)xb[tid * NN + r];
    } else if (tid < P + DE) {
      const int ci = tid - P;
      v = sEbar[0][ci] + sEbar[1][ci] + sEbar[2][ci] + sEbar[3][ci];
    } else {
      v = 0.f;
    }
    cbuf[((size_t)b * NN + r) * CP + tid] = (_Float16)v;
  }
}

// ---------------------------------------------------------------- node kernel
// fo1/fo2/fo3 over 100 nodes per batch, 16-node WMMA tiles (8 waves over 2 blocks).
template <int CP, int DO>
__global__ void __launch_bounds__(128)
node_kernel(const _Float16* __restrict__ cbuf,                  // [B,NN,CP]
            const _Float16* __restrict__ W1, const float* __restrict__ b1,
            const _Float16* __restrict__ W2, const float* __restrict__ b2,
            const _Float16* __restrict__ W3, const float* __restrict__ b3,
            _Float16* __restrict__ outh,                        // [B,DO,NN] or null
            float*    __restrict__ outf)                        // [B,DO,NN] or null
{
  constexpr int KS1 = CP / 32;
  constexpr int NT3 = (DO + 15) / 16;
  const int b = blockIdx.y;
  const int w = threadIdx.x >> 5, lane = threadIdx.x & 31;
  const int row = lane & 15, half = lane >> 4;
  const int tile = blockIdx.x * 4 + w;         // 0..7
  const int nbase = tile * 16;

  __shared__ _Float16 sH[4][16][64];
  __shared__ _Float16 sG[4][16][64];

  int node = nbase + row;
  if (node > NN - 1) node = NN - 1;            // clamp; masked at store
  const _Float16* arow = cbuf + ((size_t)b * NN + node) * CP;

  // ---- fo1: CP -> 64 (A straight from global f16, zero-padded K) ----
#pragma unroll
  for (int nt = 0; nt < 4; ++nt) {
    v8f acc = bias_frag(b1, nt * 16, 64);
#pragma unroll
    for (int k = 0; k < KS1; ++k)
      acc = wmma32(load_a_row(arow, k * 32),
                   load_b_frag(W1, 64, k * 32, nt * 16), acc);
    store_d_relu(&sH[w][0][0], nt * 16, acc);
  }
  __syncthreads();

  // ---- fo2: 64 -> 64 ----
#pragma unroll
  for (int nt = 0; nt < 4; ++nt) {
    v8f acc = bias_frag(b2, nt * 16, 64);
#pragma unroll
    for (int k = 0; k < 2; ++k)
      acc = wmma32(load_a_row(&sH[w][row][0], k * 32),
                   load_b_frag(W2, 64, k * 32, nt * 16), acc);
    store_d_relu(&sG[w][0][0], nt * 16, acc);
  }
  __syncthreads();

  // ---- fo3: 64 -> DO, relu, write transposed [B,DO,NN] ----
#pragma unroll
  for (int nt = 0; nt < NT3; ++nt) {
    v8f acc = bias_frag(b3, nt * 16, DO);
#pragma unroll
    for (int k = 0; k < 2; ++k)
      acc = wmma32(load_a_row(&sG[w][row][0], k * 32),
                   load_b_frag(W3, NT3 * 16, k * 32, nt * 16), acc);
    const int col = nt * 16 + (lane & 15);
    const int mb = half * 8;
    if (col < DO) {
#pragma unroll
      for (int v = 0; v < 8; ++v) {
        const int n2 = nbase + mb + v;
        if (n2 < NN) {
          const float val = fmaxf(acc[v], 0.f);
          if (outh) outh[((size_t)b * DO + col) * NN + n2] = (_Float16)val;
          if (outf) outf[((size_t)b * DO + col) * NN + n2] = val;
        }
      }
    }
  }
}

// ---------------------------------------------------------------- small kernels

__global__ void f32_to_f16_kernel(const float* __restrict__ in,
                                  _Float16* __restrict__ out, int n) {
  int i = blockIdx.x * blockDim.x + threadIdx.x;
  if (i < n) out[i] = (_Float16)in[i];
}

// Repack f32 W [OUT][IN] -> f16 B-matrix [KP][NP] (B = W^T, zero padded).
__global__ void convert_w_kernel(const float* __restrict__ w,
                                 _Float16* __restrict__ dst,
                                 int IN, int OUT, int KP, int NP) {
  int i = blockIdx.x * blockDim.x + threadIdx.x;
  if (i >= KP * NP) return;
  int k = i / NP, n = i % NP;
  float v = (k < IN && n < OUT) ? w[(size_t)n * IN + k] : 0.f;
  dst[i] = (_Float16)v;
}

// out[b,j] = sum_n xh[b,j,n]
__global__ void sum_nodes_kernel(const _Float16* __restrict__ xh,
                                 float* __restrict__ out, int rows) {
  int i = blockIdx.x * blockDim.x + threadIdx.x;
  if (i >= rows) return;
  const _Float16* p = xh + (size_t)i * NN;
  float s = 0.f;
  for (int n = 0; n < NN; ++n) s += (float)p[n];
  out[i] = s;
}

__global__ void linear_f32_kernel(const float* __restrict__ in,
                                  const float* __restrict__ w,
                                  const float* __restrict__ bias,
                                  float* __restrict__ out,
                                  int ROWS, int IN, int OUT, int do_relu) {
  int i = blockIdx.x * blockDim.x + threadIdx.x;
  if (i >= ROWS * OUT) return;
  int rr = i / OUT, o = i % OUT;
  const float* ip = in + (size_t)rr * IN;
  const float* wp = w + (size_t)o * IN;
  float acc = bias[o];
  for (int k = 0; k < IN; ++k) acc = fmaf(ip[k], wp[k], acc);
  if (do_relu) acc = fmaxf(acc, 0.f);
  out[i] = acc;
}

__global__ void z_kernel(const float* __restrict__ mean,
                         const float* __restrict__ logv,
                         const float* __restrict__ eps,
                         float* __restrict__ z, int n) {
  int i = blockIdx.x * blockDim.x + threadIdx.x;
  if (i < n) z[i] = mean[i] + eps[i] * expf(0.5f * logv[i]);
}

// ---------------------------------------------------------------- launch

extern "C" void kernel_launch(void* const* d_in, const int* in_sizes, int n_in,
                              void* d_out, int out_size, void* d_ws, size_t ws_size,
                              hipStream_t stream) {
  (void)in_sizes; (void)n_in; (void)out_size; (void)ws_size;

  // --- param pytree flatten order (JAX sorted dict keys) ---
  // 0:x 1:eps | dec1: 2..13 | dec2: 14..25 | dnn1: 26,27 | dnn2: 28,29 |
  // dnn3: 30,31 | enc1: 32..43 | enc2: 44..55 | logvar: 56..61 | mean: 62..67
  // inlayer order: fo1.b, fo1.w, fo2.b, fo2.w, fo3.b, fo3.w, fr1.b, fr1.w, ...
  auto F32 = [&](int i) { return (const float*)d_in[i]; };

  size_t off = 0;
  auto alloc = [&](size_t bytes) -> void* {
    off = (off + 255) & ~(size_t)255;
    void* p = (char*)d_ws + off;
    off += bytes;
    return p;
  };

  auto convW = [&](int idx, int IN, int OUT, int KP, int NP) -> const _Float16* {
    _Float16* dst = (_Float16*)alloc((size_t)KP * NP * sizeof(_Float16));
    int tot = KP * NP;
    convert_w_kernel<<<dim3((tot + 255) / 256), dim3(256), 0, stream>>>(
        F32(idx), dst, IN, OUT, KP, NP);
    return dst;
  };

  // ---- scratch buffers ----
  _Float16* xh    = (_Float16*)alloc((size_t)NB * 4 * NN * 2);
  _Float16* cbuf  = (_Float16*)alloc((size_t)NB * NN * 64 * 2);
  _Float16* enc1h = (_Float16*)alloc((size_t)NB * 10 * NN * 2);
  _Float16* enc2h = (_Float16*)alloc((size_t)NB * 20 * NN * 2);
  _Float16* dech  = (_Float16*)alloc((size_t)NB * 20 * NN * 2);
  _Float16* dec1h = (_Float16*)alloc((size_t)NB * 10 * NN * 2);
  float* Osum  = (float*)alloc((size_t)NB * 20 * 4);
  float* hA    = (float*)alloc((size_t)NB * 200 * 4);
  float* hB    = (float*)alloc((size_t)NB * 200 * 4);
  float* meanB = (float*)alloc((size_t)NB * 20 * 4);
  float* logvB = (float*)alloc((size_t)NB * 20 * 4);
  float* zB    = (float*)alloc((size_t)NB * 20 * 4);
  float* d3out = (float*)alloc((size_t)NB * 2000 * 4);

  // ---- weight repacks (f16, B-layout [Kpad][Npad]) ----
  const _Float16 *e1fr1 = convW(39,  8, 64, 32, 64);
  const _Float16 *e1fr2 = convW(41, 64, 64, 64, 64);
  const _Float16 *e1fr3 = convW(43, 64, 20, 64, 32);
  const _Float16 *e1fo1 = convW(33, 24, 64, 32, 64);
  const _Float16 *e1fo2 = convW(35, 64, 64, 64, 64);
  const _Float16 *e1fo3 = convW(37, 64, 10, 64, 16);

  const _Float16 *e2fr1 = convW(51, 20, 64, 32, 64);
  const _Float16 *e2fr2 = convW(53, 64, 64, 64, 64);
  const _Float16 *e2fr3 = convW(55, 64, 30, 64, 32);
  const _Float16 *e2fo1 = convW(45, 40, 64, 64, 64);
  const _Float16 *e2fo2 = convW(47, 64, 64, 64, 64);
  const _Float16 *e2fo3 = convW(49, 64, 20, 64, 32);

  const _Float16 *d1fr1 = convW( 9, 40, 64, 64, 64);
  const _Float16 *d1fr2 = convW(11, 64, 64, 64, 64);
  const _Float16 *d1fr3 = convW(13, 64, 30, 64, 32);
  const _Float16 *d1fo1 = convW( 3, 50, 64, 64, 64);
  const _Float16 *d1fo2 = convW( 5, 64, 64, 64, 64);
  const _Float16 *d1fo3 = convW( 7, 64, 10, 64, 16);

  const _Float16 *d2fr1 = convW(21, 20, 64, 32, 64);
  const _Float16 *d2fr2 = convW(23, 64, 64, 64, 64);
  const _Float16 *d2fr3 = convW(25, 64, 20, 64, 32);
  const _Float16 *d2fo1 = convW(15, 30, 64, 32, 64);
  const _Float16 *d2fo2 = convW(17, 64, 64, 64, 64);
  const _Float16 *d2fo3 = convW(19, 64,  4, 64, 16);

  const dim3 eg(NN, NB), ng(2, NB), blk(128);

  // x -> f16
  f32_to_f16_kernel<<<dim3((NB * 4 * NN + 255) / 256), 256, 0, stream>>>(
      F32(0), xh, NB * 4 * NN);

  // ---- encoder layer 1 : P=4, De=20, Do=10 ----
  edge_kernel<4, 20, 32, 32><<<eg, blk, 0, stream>>>(
      xh, e1fr1, F32(38), e1fr2, F32(40), e1fr3, F32(42), cbuf);
  node_kernel<32, 10><<<ng, blk, 0, stream>>>(
      cbuf, e1fo1, F32(32), e1fo2, F32(34), e1fo3, F32(36), enc1h, nullptr);

  // ---- encoder layer 2 : P=10, De=30, Do=20 ----
  edge_kernel<10, 30, 32, 64><<<eg, blk, 0, stream>>>(
      enc1h, e2fr1, F32(50), e2fr2, F32(52), e2fr3, F32(54), cbuf);
  node_kernel<64, 20><<<ng, blk, 0, stream>>>(
      cbuf, e2fo1, F32(44), e2fo2, F32(46), e2fo3, F32(48), enc2h, nullptr);

  // ---- aggregate heads, reparameterize, decoder DNN (scalar, ~0.1 GFLOP) ----
  sum_nodes_kernel<<<dim3((NB * 20 + 255) / 256), 256, 0, stream>>>(enc2h, Osum, NB * 20);

  linear_f32_kernel<<<dim3((NB * 64 + 255) / 256), 256, 0, stream>>>(Osum, F32(63), F32(62), hA, NB, 20, 64, 1);
  linear_f32_kernel<<<dim3((NB * 64 + 255) / 256), 256, 0, stream>>>(hA,   F32(65), F32(64), hB, NB, 64, 64, 1);
  linear_f32_kernel<<<dim3((NB * 20 + 255) / 256), 256, 0, stream>>>(hB,   F32(67), F32(66), meanB, NB, 64, 20, 0);

  linear_f32_kernel<<<dim3((NB * 64 + 255) / 256), 256, 0, stream>>>(Osum, F32(57), F32(56), hA, NB, 20, 64, 1);
  linear_f32_kernel<<<dim3((NB * 64 + 255) / 256), 256, 0, stream>>>(hA,   F32(59), F32(58), hB, NB, 64, 64, 1);
  linear_f32_kernel<<<dim3((NB * 20 + 255) / 256), 256, 0, stream>>>(hB,   F32(61), F32(60), logvB, NB, 64, 20, 0);

  z_kernel<<<dim3((NB * 20 + 255) / 256), 256, 0, stream>>>(
      meanB, logvB, F32(1), zB, NB * 20);

  linear_f32_kernel<<<dim3((NB * 200 + 255) / 256), 256, 0, stream>>>(zB, F32(27), F32(26), hA, NB, 20, 200, 1);
  linear_f32_kernel<<<dim3((NB * 200 + 255) / 256), 256, 0, stream>>>(hA, F32(29), F32(28), hB, NB, 200, 200, 1);
  linear_f32_kernel<<<dim3((NB * 2000 + 255) / 256), 256, 0, stream>>>(hB, F32(31), F32(30), d3out, NB, 200, 2000, 1);

  f32_to_f16_kernel<<<dim3((NB * 2000 + 255) / 256), 256, 0, stream>>>(
      d3out, dech, NB * 2000);           // reshape [B,2000]->[B,20,100] is identity

  // ---- decoder layer 1 : P=20, De=30, Do=10 ----
  edge_kernel<20, 30, 64, 64><<<eg, blk, 0, stream>>>(
      dech, d1fr1, F32(8), d1fr2, F32(10), d1fr3, F32(12), cbuf);
  node_kernel<64, 10><<<ng, blk, 0, stream>>>(
      cbuf, d1fo1, F32(2), d1fo2, F32(4), d1fo3, F32(6), dec1h, nullptr);

  // ---- decoder layer 2 : P=10, De=20, Do=4 -> f32 output ----
  edge_kernel<10, 20, 32, 32><<<eg, blk, 0, stream>>>(
      dec1h, d2fr1, F32(20), d2fr2, F32(22), d2fr3, F32(24), cbuf);
  node_kernel<32, 4><<<ng, blk, 0, stream>>>(
      cbuf, d2fo1, F32(14), d2fo2, F32(16), d2fo3, F32(18),
      nullptr, (float*)d_out);
}
